// StabilityModel_72610717106546
// MI455X (gfx1250) — matrix operand order repaired and checked by reading, not compiled, not run on hardware
//
#include <hip/hip_runtime.h>

// ============================================================================
// StabilityModel for MI455X (gfx1250, wave32, WMMA).
//
// Roofline:
//  * pair[B,L,L,64] (268MB/branch) is never materialized; the fused attention
//    kernel recomputes pair_t[b,l,m,:] in registers from pair3[b,l,m,:3]
//    through the (<=2)-step LN chain using tiny pl/pr[B,L,64] projections.
//    HBM traffic ~0.2GB -> ~10us floor at 23.3TB/s.
//  * All dense layers use v_wmma_f32_16x16x32_f16. Weights are pre-packed
//    (transposed, zero-padded, f16) so fragments load as contiguous b128s
//    with zero bounds checks inside the K loop.
// ============================================================================

typedef __attribute__((ext_vector_type(16))) _Float16 v16h;
typedef __attribute__((ext_vector_type(8)))  _Float16 v8h;
typedef __attribute__((ext_vector_type(8)))  float    v8f;

#define ACT_NONE  0
#define ACT_LRELU 1
#define MROWS 2048      // B*L = 4*512
#define LSEQ  512
#define BDIM  4

__device__ __forceinline__ float warp_sum(float v) {
  #pragma unroll
  for (int o = 16; o > 0; o >>= 1) v += __shfl_xor(v, o, 32);
  return v;
}
__device__ __forceinline__ float warp_max(float v) {
  #pragma unroll
  for (int o = 16; o > 0; o >>= 1) v = fmaxf(v, __shfl_xor(v, o, 32));
  return v;
}

// ---------------------------------------------------------------------------
// Weight pack: WT[Np][Kp] (f16, zero pad) = W[K][N] transposed.
// ---------------------------------------------------------------------------
__global__ void pack_w_kernel(const float* __restrict__ W, int K, int N,
                              _Float16* __restrict__ WT, int Kp, int Np) {
  const size_t idx = (size_t)blockIdx.x * blockDim.x + threadIdx.x;
  const size_t total = (size_t)Kp * Np;
  if (idx >= total) return;
  const int n = (int)(idx / Kp);
  const int k = (int)(idx % Kp);
  WT[idx] = (n < N && k < K) ? (_Float16)W[(size_t)k * N + n] : (_Float16)0.f;
}

// ---------------------------------------------------------------------------
// Row LayerNorm over K (K up to 1280). One block per row.
// ---------------------------------------------------------------------------
__global__ void ln_rows_kernel(const float* __restrict__ X,
                               const float* __restrict__ g,
                               const float* __restrict__ b,
                               float* __restrict__ Y, int K) {
  const int row = blockIdx.x;
  const float* x = X + (size_t)row * K;
  float* y = Y + (size_t)row * K;
  float s = 0.f, s2 = 0.f;
  for (int i = threadIdx.x; i < K; i += blockDim.x) {
    float v = x[i]; s += v; s2 += v * v;
  }
  __shared__ float red0[32], red1[32];
  s = warp_sum(s); s2 = warp_sum(s2);
  const int wave = threadIdx.x >> 5, lane = threadIdx.x & 31;
  if (lane == 0) { red0[wave] = s; red1[wave] = s2; }
  __syncthreads();
  const int nw = blockDim.x >> 5;
  if (wave == 0) {
    float a = (lane < nw) ? red0[lane] : 0.f;
    float c = (lane < nw) ? red1[lane] : 0.f;
    a = warp_sum(a); c = warp_sum(c);
    if (lane == 0) { red0[0] = a; red1[0] = c; }
  }
  __syncthreads();
  const float mean = red0[0] / (float)K;
  const float var  = red1[0] / (float)K - mean * mean;
  const float inv  = rsqrtf(var + 1e-5f);
  for (int i = threadIdx.x; i < K; i += blockDim.x)
    y[i] = (x[i] - mean) * inv * g[i] + b[i];
}

// ---------------------------------------------------------------------------
// WMMA GEMM: C[M,Np] = act(A[M,Kp] @ WT^T + bias).
// A f32 stride lda (rows padded: values at k in [K,Kp) multiply zero WT pad).
// WT f16 [Np][Kp] packed. No guards inside K loop; per the ISA 7.12.2
// wave32 fragment layouts. 256 thr = 8 waves, block tile 64x32, wave 16x16.
// ---------------------------------------------------------------------------
__global__ void gemm_wmma_kernel(const float* __restrict__ A, int lda,
                                 const _Float16* __restrict__ WT, int Kp, int Np,
                                 const float* __restrict__ bias, int Nreal,
                                 float* __restrict__ C, int ldc, int act) {
  const int wave = threadIdx.x >> 5, lane = threadIdx.x & 31;
  const int row0 = blockIdx.y * 64 + (wave & 3) * 16;
  const int col0 = blockIdx.x * 32 + (wave >> 2) * 16;
  if (col0 >= Np) return;           // wave-uniform early out (no __syncthreads)
  const int half = lane >> 4;       // lane 0-15 vs 16-31
  const int l15  = lane & 15;
  const int arow = row0 + l15;      // A row held by this lane
  const int bcol = col0 + l15;      // B/C column held by this lane (< Np)

  v8f acc;
  const float bv = (bias != nullptr && bcol < Nreal) ? bias[bcol] : 0.f;
  #pragma unroll
  for (int j = 0; j < 8; j++) acc[j] = bv;

  const float* ap = A + (size_t)arow * lda;
  const _Float16* wp = WT + (size_t)bcol * Kp;

  for (int k0 = 0; k0 < Kp; k0 += 32) {
    // A fragment: half 0 -> K = k0+{0..7,16..23}; half 1 -> +8.
    const int abase = k0 + half * 8;
    const float4 a0 = *(const float4*)(ap + abase);
    const float4 a1 = *(const float4*)(ap + abase + 4);
    const float4 a2 = *(const float4*)(ap + abase + 16);
    const float4 a3 = *(const float4*)(ap + abase + 20);
    v16h af;
    af[0] = (_Float16)a0.x;  af[1] = (_Float16)a0.y;
    af[2] = (_Float16)a0.z;  af[3] = (_Float16)a0.w;
    af[4] = (_Float16)a1.x;  af[5] = (_Float16)a1.y;
    af[6] = (_Float16)a1.z;  af[7] = (_Float16)a1.w;
    af[8] = (_Float16)a2.x;  af[9] = (_Float16)a2.y;
    af[10] = (_Float16)a2.z; af[11] = (_Float16)a2.w;
    af[12] = (_Float16)a3.x; af[13] = (_Float16)a3.y;
    af[14] = (_Float16)a3.z; af[15] = (_Float16)a3.w;
    // B fragment: half h holds K = k0 + h*8 + {0..7, 16..23} for its column.
    const v8h b0 = *(const v8h*)(wp + k0 + half * 8);
    const v8h b1 = *(const v8h*)(wp + k0 + half * 8 + 16);
    v16h bf;
    #pragma unroll
    for (int e = 0; e < 8; e++) { bf[e] = b0[e]; bf[e + 8] = b1[e]; }

    acc = __builtin_amdgcn_wmma_f32_16x16x32_f16(
        false, af, false, bf, (short)0, acc, false, false);
  }

  #pragma unroll
  for (int j = 0; j < 8; j++) {     // C: M = j + 8*half, N = lane&15
    float v = acc[j];
    if (act == ACT_LRELU) v = (v > 0.f) ? v : v * 0.01f;
    C[(size_t)(row0 + j + half * 8) * ldc + bcol] = v;
  }
}

// ---------------------------------------------------------------------------
// x = LN(x + r) over dim 32 (in place). 8 rows per 256-thread block.
// ---------------------------------------------------------------------------
__global__ void add_ln32_kernel(float* __restrict__ X, const float* __restrict__ R,
                                const float* __restrict__ g, const float* __restrict__ b,
                                int M) {
  const int row = blockIdx.x * 8 + (threadIdx.x >> 5);
  const int c = threadIdx.x & 31;
  if (row >= M) return;
  float v = X[(size_t)row * 32 + c] + R[(size_t)row * 32 + c];
  const float mean = warp_sum(v) * (1.f / 32.f);
  const float d = v - mean;
  const float var = warp_sum(d * d) * (1.f / 32.f);
  X[(size_t)row * 32 + c] = d * rsqrtf(var + 1e-5f) * g[c] + b[c];
}

// ---------------------------------------------------------------------------
// Fused attention for one (b,l): logits = q.k/2 + pair_bias + plddt_bias,
// softmax over m, o = attn @ v. pair_bias recomputed from pair3 through the
// <=2-step LN chain (nsteps = block index). 512 threads = 16 waves, m = tid.
// ---------------------------------------------------------------------------
__global__ void attention_kernel(
    const float* __restrict__ Q, const float* __restrict__ Kb, const float* __restrict__ V,
    const float* __restrict__ plddt,   // [B,L]
    const float* __restrict__ pair3,   // [B,L,L,3]
    const float* __restrict__ pe_w, const float* __restrict__ pe_b,  // [3,64],[64]
    const float* __restrict__ pl1, const float* __restrict__ pr1,    // [B,L,64]
    const float* __restrict__ g1,  const float* __restrict__ b1,
    const float* __restrict__ pl2, const float* __restrict__ pr2,
    const float* __restrict__ g2,  const float* __restrict__ b2,
    int nsteps,
    const float* __restrict__ wpb, const float* __restrict__ bpb,    // [64,8],[8]
    const float* __restrict__ wpl, const float* __restrict__ bpl,    // [8],[8]
    float* __restrict__ O, int L) {
  const int l = blockIdx.x, bi = blockIdx.y;
  const int tid = threadIdx.x;
  const size_t rowql = (size_t)(bi * L + l);

  __shared__ float sq[32];
  __shared__ float s_pew[192], s_peb[64], s_wpb[512], s_bpb[8], s_wpl[8], s_bpl[8];
  __shared__ float s_pl1[64], s_g1[64], s_b1[64];
  __shared__ float s_pl2[64], s_g2[64], s_b2[64];
  __shared__ float red[128];     // [h=8][wave=16]
  __shared__ float ored[512];    // [wave=16][c=32]
  __shared__ float s_hv[8];

  for (int i = tid; i < 192; i += 512) s_pew[i] = pe_w[i];
  for (int i = tid; i < 64;  i += 512) s_peb[i] = pe_b[i];
  for (int i = tid; i < 512; i += 512) s_wpb[i] = wpb[i];
  if (tid < 8) { s_bpb[tid] = bpb[tid]; s_wpl[tid] = wpl[tid]; s_bpl[tid] = bpl[tid]; }
  if (tid < 32) sq[tid] = Q[rowql * 32 + tid];
  if (nsteps >= 1)
    for (int i = tid; i < 64; i += 512) { s_pl1[i] = pl1[rowql * 64 + i]; s_g1[i] = g1[i]; s_b1[i] = b1[i]; }
  if (nsteps >= 2)
    for (int i = tid; i < 64; i += 512) { s_pl2[i] = pl2[rowql * 64 + i]; s_g2[i] = g2[i]; s_b2[i] = b2[i]; }
  __syncthreads();

  const int m = tid;                     // blockDim.x == L == 512
  const size_t rowm = (size_t)(bi * L + m);

  // q.k / sqrt(hd=4) + plddt bias (key side)
  float lg[8];
  {
    const float* krow = Kb + rowm * 32;
    const float pld = plddt[rowm];
    #pragma unroll
    for (int h = 0; h < 8; h++) {
      float s = 0.f;
      #pragma unroll
      for (int d = 0; d < 4; d++) s += sq[h * 4 + d] * krow[h * 4 + d];
      lg[h] = s * 0.5f + pld * s_wpl[h] + s_bpl[h];
    }
  }

  // pair chain recompute: p0 = pair3 @ pe_w + pe_b, then LN chain steps
  float p[64];
  {
    const float* p3 = pair3 + (rowql * L + m) * 3;
    const float a0 = p3[0], a1 = p3[1], a2 = p3[2];
    #pragma unroll
    for (int c = 0; c < 64; c++)
      p[c] = s_peb[c] + a0 * s_pew[c] + a1 * s_pew[64 + c] + a2 * s_pew[128 + c];
  }
  if (nsteps >= 1) {
    const float* prm = pr1 + rowm * 64;
    float s = 0.f, s2 = 0.f;
    #pragma unroll
    for (int c = 0; c < 64; c++) { p[c] += s_pl1[c] + prm[c]; s += p[c]; }
    const float mean = s * (1.f / 64.f);
    #pragma unroll
    for (int c = 0; c < 64; c++) { const float d = p[c] - mean; s2 += d * d; }
    const float inv = rsqrtf(s2 * (1.f / 64.f) + 1e-5f);
    #pragma unroll
    for (int c = 0; c < 64; c++) p[c] = (p[c] - mean) * inv * s_g1[c] + s_b1[c];
  }
  if (nsteps >= 2) {
    const float* prm = pr2 + rowm * 64;
    float s = 0.f, s2 = 0.f;
    #pragma unroll
    for (int c = 0; c < 64; c++) { p[c] += s_pl2[c] + prm[c]; s += p[c]; }
    const float mean = s * (1.f / 64.f);
    #pragma unroll
    for (int c = 0; c < 64; c++) { const float d = p[c] - mean; s2 += d * d; }
    const float inv = rsqrtf(s2 * (1.f / 64.f) + 1e-5f);
    #pragma unroll
    for (int c = 0; c < 64; c++) p[c] = (p[c] - mean) * inv * s_g2[c] + s_b2[c];
  }
  // pair bias projection [64]->[8]
  #pragma unroll
  for (int h = 0; h < 8; h++) {
    float s = s_bpb[h];
    #pragma unroll
    for (int c = 0; c < 64; c++) s += p[c] * s_wpb[c * 8 + h];
    lg[h] += s;
  }

  // softmax over m (512 threads) per head
  const int wave = tid >> 5, lane = tid & 31;
  #pragma unroll
  for (int h = 0; h < 8; h++) { float v = warp_max(lg[h]); if (lane == 0) red[h * 16 + wave] = v; }
  __syncthreads();
  if (tid < 8) {
    float v = -3.4e38f;
    for (int w = 0; w < 16; w++) v = fmaxf(v, red[tid * 16 + w]);
    s_hv[tid] = v;
  }
  __syncthreads();
  #pragma unroll
  for (int h = 0; h < 8; h++) lg[h] = __expf(lg[h] - s_hv[h]);
  __syncthreads();
  #pragma unroll
  for (int h = 0; h < 8; h++) { float v = warp_sum(lg[h]); if (lane == 0) red[h * 16 + wave] = v; }
  __syncthreads();
  if (tid < 8) {
    float v = 0.f;
    for (int w = 0; w < 16; w++) v += red[tid * 16 + w];
    s_hv[tid] = v;
  }
  __syncthreads();
  #pragma unroll
  for (int h = 0; h < 8; h++) lg[h] /= s_hv[h];

  // o[c] = sum_m attn[c>>2] * v[m][c]
  float oacc[32];
  {
    const float* vrow = V + rowm * 32;
    #pragma unroll
    for (int c = 0; c < 32; c++) oacc[c] = lg[c >> 2] * vrow[c];
  }
  #pragma unroll
  for (int c = 0; c < 32; c++) {
    float v = warp_sum(oacc[c]);
    if (lane == 0) ored[wave * 32 + c] = v;
  }
  __syncthreads();
  if (tid < 32) {
    float v = 0.f;
    for (int w = 0; w < 16; w++) v += ored[w * 32 + tid];
    O[rowql * 32 + tid] = v;
  }
}

// ---------------------------------------------------------------------------
// Copy [M,N] f32 block between strided matrices (readout concat tail).
// ---------------------------------------------------------------------------
__global__ void copy_mat_kernel(const float* __restrict__ src, int lds,
                                float* __restrict__ dst, int ldd, int M, int N) {
  const int idx = blockIdx.x * blockDim.x + threadIdx.x;
  const int r = idx / N, c = idx % N;
  if (r < M) dst[(size_t)r * ldd + c] = src[(size_t)r * lds + c];
}

// ---------------------------------------------------------------------------
// out[b] = coef * sum_l (mut_val - wt_val)[b,l] * (wt_tok != mut_tok) + cons
// val buffers have stride VSTRIDE (padded N from the WMMA GEMM).
// ---------------------------------------------------------------------------
#define VSTRIDE 16
__global__ void final_kernel(const float* __restrict__ wt_val, const float* __restrict__ mut_val,
                             const int* __restrict__ wt_tok, const int* __restrict__ mut_tok,
                             const float* __restrict__ coef, const float* __restrict__ cons,
                             float* __restrict__ out, int L) {
  const int b = blockIdx.x;
  float s = 0.f;
  for (int i = threadIdx.x; i < L; i += blockDim.x) {
    const int idx = b * L + i;
    if (wt_tok[idx] != mut_tok[idx])
      s += mut_val[(size_t)idx * VSTRIDE] - wt_val[(size_t)idx * VSTRIDE];
  }
  __shared__ float red[32];
  s = warp_sum(s);
  const int wave = threadIdx.x >> 5, lane = threadIdx.x & 31;
  if (lane == 0) red[wave] = s;
  __syncthreads();
  if (threadIdx.x == 0) {
    float t = 0.f;
    const int nw = blockDim.x >> 5;
    for (int w = 0; w < nw; w++) t += red[w];
    out[b] = coef[0] * t + cons[0];
  }
}

// ============================================================================
// Host orchestration.
//
// Input index map. Top-level: setup_inputs() insertion order:
//   0 wt_embedding  1 wt_pair  2 wt_plddt  3 wt_tokens
//   4 mut_embedding 5 mut_pair 6 mut_plddt 7 mut_tokens
// 'params' flattened jax-style (dict keys alphabetical, lists in order):
//   blocks[t] (t=0..2), 24 leaves each alphabetical:
//     bk,bo,bpb,bpl,bq,bv,fb1,fb2,fw1,fw2,ln1b,ln1g,ln2b,ln2g,
//     plnb,plng,pwl,pwr,wk,wo,wpb,wpl,wq,wv          -> 8 + 24*t + j
//   coef=80 cons=81
//   ds: b1,b2,b3,lnb,lng,r1b,r1w,r2b,r2w,r3b,r3w,w1,w2,w3 -> 82..95
//   esm: b1,b2,b3,b4,lnb,lng,w1,w2,w3,w4                  -> 96..105
//   pe_b=106 pe_w=107
// ============================================================================
enum { BK=0,BO,BPB,BPL,BQ,BV,FB1,FB2,FW1,FW2,LN1B,LN1G,LN2B,LN2G,
       PLNB,PLNG,PWL,PWR,WK,WO,WPB,WPL,WQ,WV };
enum { DS_B1=82,DS_B2,DS_B3,DS_LNB,DS_LNG,DS_R1B,DS_R1W,DS_R2B,DS_R2W,
       DS_R3B,DS_R3W,DS_W1,DS_W2,DS_W3 };
enum { ESM_B1=96,ESM_B2,ESM_B3,ESM_B4,ESM_LNB,ESM_LNG,ESM_W1,ESM_W2,ESM_W3,ESM_W4 };
#define P_COEF 80
#define P_CONS 81
#define P_PEB 106
#define P_PEW 107

struct PackedW { const _Float16* wt; int Kp, Np; };

extern "C" void kernel_launch(void* const* d_in, const int* in_sizes, int n_in,
                              void* d_out, int out_size, void* d_ws, size_t ws_size,
                              hipStream_t stream) {
  (void)in_sizes; (void)n_in; (void)out_size; (void)ws_size;
  auto P = [&](int i) -> const float* { return (const float*)d_in[i]; };
  auto Bp = [&](int t, int j) -> const float* { return (const float*)d_in[8 + 24 * t + j]; };

  const float* wt_E   = P(0); const float* wt_pair = P(1);
  const float* wt_pld = P(2); const int*   wt_tok  = (const int*)d_in[3];
  const float* mu_E   = P(4); const float* mu_pair = P(5);
  const float* mu_pld = P(6); const int*   mu_tok  = (const int*)d_in[7];

  // --- f32 workspace (all offsets multiples of 16 floats) -------------------
  float* w = (float*)d_ws;
  size_t off = 0;
  auto alloc = [&](size_t nf) { float* p = w + off; off += (nf + 15) & ~(size_t)15; return p; };
  float* LN_OUT = alloc((size_t)MROWS * 1280);
  float* H1     = alloc((size_t)MROWS * 640);
  float* H2     = alloc((size_t)MROWS * 320);
  float* H3     = alloc((size_t)MROWS * 32);
  float* X      = alloc((size_t)MROWS * 32);
  float* TMP    = alloc((size_t)MROWS * 32);
  float* F1     = alloc((size_t)MROWS * 64);
  float* Qb     = alloc((size_t)MROWS * 32);
  float* Kbuf   = alloc((size_t)MROWS * 32);
  float* Vb     = alloc((size_t)MROWS * 32);
  float* Ob     = alloc((size_t)MROWS * 32);
  float* PL1    = alloc((size_t)MROWS * 64);
  float* PR1    = alloc((size_t)MROWS * 64);
  float* PL2    = alloc((size_t)MROWS * 64);
  float* PR2    = alloc((size_t)MROWS * 64);
  float* CAT    = alloc((size_t)MROWS * 160 + 32);  // slack for K-pad reads
  float* R1     = alloc((size_t)MROWS * 80 + 32);   // stride 80, Kp=96 reads
  float* R2     = alloc((size_t)MROWS * 48 + 32);   // stride 48, Kp=64 reads
  float* VALWT  = alloc((size_t)MROWS * VSTRIDE);
  float* VALMUT = alloc((size_t)MROWS * VSTRIDE);

  // --- packed f16 weights ---------------------------------------------------
  _Float16* hbase = (_Float16*)(w + off);
  size_t hoff = 0;
  auto pack = [&](int pidx, int K, int N) -> PackedW {
    const int Kp = (K + 31) & ~31, Np = (N + 15) & ~15;
    _Float16* dst = hbase + hoff;
    const size_t n = (size_t)Kp * Np;
    hoff += (n + 31) & ~(size_t)31;                 // keep 64B alignment
    pack_w_kernel<<<(int)((n + 255) / 256), 256, 0, stream>>>(P(pidx), K, N, dst, Kp, Np);
    return {dst, Kp, Np};
  };

  PackedW eW1 = pack(ESM_W1, 1280, 640), eW2 = pack(ESM_W2, 640, 320);
  PackedW eW3 = pack(ESM_W3, 320, 32),   eW4 = pack(ESM_W4, 32, 32);
  PackedW dW1 = pack(DS_W1, 1280, 640),  dW2 = pack(DS_W2, 640, 320);
  PackedW dW3 = pack(DS_W3, 320, 128);
  PackedW r1W = pack(DS_R1W, 160, 80),   r2W = pack(DS_R2W, 80, 40);
  PackedW r3W = pack(DS_R3W, 40, 1);
  PackedW bWQ[3], bWK[3], bWV[3], bWO[3], bFW1[3], bFW2[3], bPWL[3], bPWR[3];
  for (int t = 0; t < 3; t++) {
    bWQ[t] = pack(8 + 24 * t + WQ, 32, 32);
    bWK[t] = pack(8 + 24 * t + WK, 32, 32);
    bWV[t] = pack(8 + 24 * t + WV, 32, 32);
    bWO[t] = pack(8 + 24 * t + WO, 32, 32);
    bFW1[t] = pack(8 + 24 * t + FW1, 32, 64);
    bFW2[t] = pack(8 + 24 * t + FW2, 64, 32);
    bPWL[t] = pack(8 + 24 * t + PWL, 32, 64);
    bPWR[t] = pack(8 + 24 * t + PWR, 32, 64);
  }

  auto G = [&](const float* A, int lda, PackedW pw, const float* bias, int Nreal,
               float* C, int ldc, int act) {
    dim3 g((pw.Np + 31) / 32, MROWS / 64);
    gemm_wmma_kernel<<<g, 256, 0, stream>>>(A, lda, pw.wt, pw.Kp, pw.Np,
                                            bias, Nreal, C, ldc, act);
  };

  auto run_branch = [&](const float* E, const float* pair3, const float* pld,
                        float* val_out) {
    // --- ESM transform: LN -> 640 -> 320 -> 32 -> 32
    ln_rows_kernel<<<MROWS, 256, 0, stream>>>(E, P(ESM_LNG), P(ESM_LNB), LN_OUT, 1280);
    G(LN_OUT, 1280, eW1, P(ESM_B1), 640, H1, 640, ACT_LRELU);
    G(H1, 640, eW2, P(ESM_B2), 320, H2, 320, ACT_LRELU);
    G(H2, 320, eW3, P(ESM_B3), 32, H3, 32, ACT_LRELU);
    G(H3, 32, eW4, P(ESM_B4), 32, X, 32, ACT_NONE);

    // --- 3 GAT blocks
    for (int t = 0; t < 3; t++) {
      G(X, 32, bWQ[t], Bp(t, BQ), 32, Qb, 32, ACT_NONE);
      G(X, 32, bWK[t], Bp(t, BK), 32, Kbuf, 32, ACT_NONE);
      G(X, 32, bWV[t], Bp(t, BV), 32, Vb, 32, ACT_NONE);
      attention_kernel<<<dim3(LSEQ, BDIM), 512, 0, stream>>>(
          Qb, Kbuf, Vb, pld, pair3, P(P_PEW), P(P_PEB),
          PL1, PR1, Bp(0, PLNG), Bp(0, PLNB),
          PL2, PR2, Bp(1, PLNG), Bp(1, PLNB),
          /*nsteps=*/t,
          Bp(t, WPB), Bp(t, BPB), Bp(t, WPL), Bp(t, BPL), Ob, LSEQ);
      G(Ob, 32, bWO[t], Bp(t, BO), 32, TMP, 32, ACT_NONE);
      add_ln32_kernel<<<MROWS / 8, 256, 0, stream>>>(X, TMP, Bp(t, LN1G), Bp(t, LN1B), MROWS);
      G(X, 32, bFW1[t], Bp(t, FB1), 64, F1, 64, ACT_LRELU);
      G(F1, 64, bFW2[t], Bp(t, FB2), 32, TMP, 32, ACT_NONE);
      add_ln32_kernel<<<MROWS / 8, 256, 0, stream>>>(X, TMP, Bp(t, LN2G), Bp(t, LN2B), MROWS);
      if (t < 2) {  // pair-update vectors for later blocks' LN chain
        G(X, 32, bPWL[t], nullptr, 64, (t == 0) ? PL1 : PL2, 64, ACT_NONE);
        G(X, 32, bPWR[t], nullptr, 64, (t == 0) ? PR1 : PR2, 64, ACT_NONE);
      }
    }

    // --- DS transform into CAT cols [0,128), node x into cols [128,160)
    ln_rows_kernel<<<MROWS, 256, 0, stream>>>(E, P(DS_LNG), P(DS_LNB), LN_OUT, 1280);
    G(LN_OUT, 1280, dW1, P(DS_B1), 640, H1, 640, ACT_LRELU);
    G(H1, 640, dW2, P(DS_B2), 320, H2, 320, ACT_LRELU);
    G(H2, 320, dW3, P(DS_B3), 128, CAT, 160, ACT_NONE);
    copy_mat_kernel<<<(MROWS * 32 + 255) / 256, 256, 0, stream>>>(X, 32, CAT + 128, 160, MROWS, 32);

    // --- Readout 160 -> 80 -> 40 -> 1 (padded strides 80 / 48 / 16)
    G(CAT, 160, r1W, P(DS_R1B), 80, R1, 80, ACT_LRELU);
    G(R1, 80, r2W, P(DS_R2B), 40, R2, 48, ACT_LRELU);
    G(R2, 48, r3W, P(DS_R3B), 1, val_out, VSTRIDE, ACT_NONE);
  };

  run_branch(wt_E, wt_pair, wt_pld, VALWT);
  run_branch(mu_E, mu_pair, mu_pld, VALMUT);

  final_kernel<<<BDIM, 256, 0, stream>>>(VALWT, VALMUT, wt_tok, mu_tok,
                                         P(P_COEF), P(P_CONS), (float*)d_out, LSEQ);
}
